// DeltaNet_31877247271455
// MI455X (gfx1250) — matrix operand order repaired and checked by reading, not compiled, not run on hardware
//
#include <hip/hip_runtime.h>
#include <hip/hip_bf16.h>
#include <math.h>

// ---------------- CDNA5 WMMA types ----------------
typedef __attribute__((ext_vector_type(16))) __bf16 v16bf;
typedef __attribute__((ext_vector_type(8)))  __bf16 v8bf;
typedef __attribute__((ext_vector_type(8)))  float  v8f;
typedef __attribute__((ext_vector_type(4)))  float  v4f;
typedef __attribute__((ext_vector_type(2)))  float  v2f;

#define DEV static __device__ __forceinline__

DEV float sigmoidf_(float x) { return 1.f / (1.f + __expf(-x)); }
DEV float siluf_(float x)    { return x * sigmoidf_(x); }

// fp32 matrix pipe: D(16x16) = A(16x4) * B(4x16) + C
DEV v8f wmma_f32x4(v2f a, v2f b, v8f c) {
#if __has_builtin(__builtin_amdgcn_wmma_f32_16x16x4_f32)
  return __builtin_amdgcn_wmma_f32_16x16x4_f32(false, a, false, b, (short)0, c, false, false);
#else
  c[0] += a[0] * b[0];  // placeholder; not expected on gfx1250
  return c;
#endif
}

// bf16 matrix pipe: D(16x16,f32) = A(16x32,bf16) * B(32x16,bf16) + C
DEV v8f wmma_bf16_(v16bf a, v16bf b, v8f c) {
  return __builtin_amdgcn_wmma_f32_16x16x32_bf16(false, a, false, b, (short)0, c, false, false);
}

DEV v16bf make16(v8bf lo, v8bf hi) {
  v16bf x;
#pragma unroll
  for (int e = 0; e < 8; ++e) { x[e] = lo[e]; x[8 + e] = hi[e]; }
  return x;
}

// ---------------- model constants ----------------
#define BB     4
#define LL     2048
#define DD     1024
#define NHEADS 4
#define DKH    256      // head dim (dk == dv)
#define CH     32       // chunk
#define DVS    32       // dv slice per block (independent S columns)
#define KCONV  4
#define LDQ    260      // padded LDS stride for 32x256 tiles (bank-conflict free)
#define LDT    36       // padded LDS stride for 32-col tiles

// ---------------- f32 -> bf16 convert (8 elems / thread, b128 traffic) ----------------
__global__ void cvt_bf16_v8(const float* __restrict__ x, __bf16* __restrict__ y, int n) {
  int i = (blockIdx.x * blockDim.x + threadIdx.x) * 8;
  if (i >= n) return;
  v4f a = *(const v4f*)&x[i];
  v4f b = *(const v4f*)&x[i + 4];
  v8bf o;
#pragma unroll
  for (int e = 0; e < 4; ++e) { o[e] = (__bf16)a[e]; o[4 + e] = (__bf16)b[e]; }
  *(v8bf*)&y[i] = o;
}

// ---------------- f32 [K,N] -> bf16 transposed [N,K] ----------------
__global__ void cvt_transpose_bf16(const float* __restrict__ W, __bf16* __restrict__ T,
                                   int K, int N) {
  int idx = blockIdx.x * blockDim.x + threadIdx.x;
  if (idx >= K * N) return;
  int k = idx / N, n = idx % N;
  T[(size_t)n * K + k] = (__bf16)W[(size_t)k * N + n];
}

// ---------------- bf16 WMMA GEMM: C[M,N] = A[M,K] @ Bt[N,K]^T ----------------
// block = 8 waves; wave tile 16x64 (4 accumulators); block tile 32x256.
// A row-major [M,K]; Bt holds B transposed [N,K] so every fragment is 2x b128.
// Software-pipelined: fragments for k0+32 are in flight while WMMAs consume k0.
__global__ __launch_bounds__(256)
void gemm_bf16_wmma(const __bf16* __restrict__ A, const __bf16* __restrict__ Bt,
                    float* __restrict__ C, int M, int N, int K) {
  const int tid  = threadIdx.x;
  const int wave = tid >> 5, lane = tid & 31;
  const int r    = lane & 15, half = lane >> 4;
  const int row0 = blockIdx.y * 32  + (wave >> 2) * 16;
  const int col0 = blockIdx.x * 256 + (wave & 3)  * 64;

  const v8f vz = {0.f,0.f,0.f,0.f,0.f,0.f,0.f,0.f};
  v8f acc[4];
#pragma unroll
  for (int t = 0; t < 4; ++t) acc[t] = vz;

  const __bf16* arow = A + (size_t)(row0 + r) * K + half * 8;
  const __bf16* brow[4];
#pragma unroll
  for (int t = 0; t < 4; ++t)
    brow[t] = Bt + (size_t)(col0 + t * 16 + r) * K + half * 8;

  // prologue: fragments for k0 = 0
  v16bf a_cur = make16(*(const v8bf*)(arow), *(const v8bf*)(arow + 16));
  v16bf b_cur[4];
#pragma unroll
  for (int t = 0; t < 4; ++t)
    b_cur[t] = make16(*(const v8bf*)(brow[t]), *(const v8bf*)(brow[t] + 16));

  for (int k0 = 32; k0 < K; k0 += 32) {
    // issue next tile's loads before consuming current fragments
    v16bf a_nxt = make16(*(const v8bf*)(arow + k0), *(const v8bf*)(arow + k0 + 16));
    v16bf b_nxt[4];
#pragma unroll
    for (int t = 0; t < 4; ++t)
      b_nxt[t] = make16(*(const v8bf*)(brow[t] + k0), *(const v8bf*)(brow[t] + k0 + 16));
    __builtin_prefetch(arow + k0 + 32, 0, 1);
#pragma unroll
    for (int t = 0; t < 4; ++t)
      __builtin_prefetch(brow[t] + k0 + 32, 0, 1);

#pragma unroll
    for (int t = 0; t < 4; ++t)
      acc[t] = wmma_bf16_(a_cur, b_cur[t], acc[t]);

    a_cur = a_nxt;
#pragma unroll
    for (int t = 0; t < 4; ++t) b_cur[t] = b_nxt[t];
  }
  // epilogue
#pragma unroll
  for (int t = 0; t < 4; ++t)
    acc[t] = wmma_bf16_(a_cur, b_cur[t], acc[t]);

#pragma unroll
  for (int t = 0; t < 4; ++t)
#pragma unroll
    for (int j = 0; j < 8; ++j)
      C[(size_t)(row0 + half * 8 + j) * N + col0 + t * 16 + r] = acc[t][j];
}

// ---------------- small projections: beta/gamma/mix_lin (N=H=4 each) ----------------
__global__ void proj_small(const float* __restrict__ x, const float* __restrict__ Wb,
                           const float* __restrict__ Wdec, const float* __restrict__ Wmix,
                           float* __restrict__ beta, float* __restrict__ gamma,
                           float* __restrict__ mixlin, int rows, int K, int H) {
  int idx = blockIdx.x * blockDim.x + threadIdx.x;
  if (idx >= rows * H * 3) return;
  int which = idx / (rows * H);
  int rem   = idx % (rows * H);
  int row = rem / H, h = rem % H;
  const float* W  = (which == 0) ? Wb : (which == 1) ? Wdec : Wmix;
  const float* xr = x + (size_t)row * K;
  float acc = 0.f;
  for (int kk = 0; kk < K; ++kk) acc += xr[kk] * W[kk * H + h];
  if      (which == 0) beta  [row * H + h] = sigmoidf_(acc);
  else if (which == 1) gamma [row * H + h] = sigmoidf_(acc);
  else                 mixlin[row * H + h] = acc;
}

// ---------------- depthwise short conv (K=4, causal) + silu^n ----------------
// x has row-stride ldx and column offset xoff (to read a slice of the fused GEMM out)
__global__ void conv_silu(const float* __restrict__ x, const float* __restrict__ w,
                          float* __restrict__ y, int B, int L, int C, int ldx, int xoff,
                          int nsilu) {
  int idx = blockIdx.x * blockDim.x + threadIdx.x;
  if (idx >= B * L * C) return;
  int c = idx % C, t = (idx / C) % L, b = idx / (C * L);
  float acc = 0.f;
#pragma unroll
  for (int k = 0; k < KCONV; ++k) {
    int tt = t - (KCONV - 1) + k;
    if (tt >= 0) acc += x[((size_t)b * L + tt) * ldx + xoff + c] * w[c * KCONV + k];
  }
  for (int s = 0; s < nsilu; ++s) acc = siluf_(acc);
  y[idx] = acc;
}

// ---------------- chunkwise delta rule (fp32 WMMA, S slice in LDS) ----------------
__global__ __launch_bounds__(256)
void delta_kernel(const float* __restrict__ qg, const float* __restrict__ kg,
                  const float* __restrict__ vg, const float* __restrict__ betag,
                  float* __restrict__ og, int B, int L) {
  extern __shared__ float sm[];
  float* sQ  = sm;                  // 32 x LDQ
  float* sK  = sQ  + CH  * LDQ;
  float* sKB = sK  + CH  * LDQ;
  float* sW  = sKB + CH  * LDQ;
  float* sS  = sW  + CH  * LDQ;     // 256 x LDT (fp32 state slice)
  float* sV  = sS  + DKH * LDT;     // 32 x LDT
  float* sU  = sV  + CH  * LDT;
  float* sT  = sU  + CH  * LDT;
  float* sAi = sT  + CH  * LDT;
  float* sBt = sAi + CH  * LDT;     // 32
  float* sNq = sBt + CH;            // 32
  float* sNk = sNq + CH;            // 32

  const int tid  = threadIdx.x;
  const int wave = tid >> 5, lane = tid & 31;
  const int r    = lane & 15, half = lane >> 4;

  const int nslice = DKH / DVS;
  const int bh = blockIdx.x / nslice, sl = blockIdx.x % nslice;
  const int b_i = bh / NHEADS, h_i = bh % NHEADS;
  const int D = NHEADS * DKH;
  const size_t rowBase = (size_t)b_i * L;
  const int colQK = h_i * DKH;
  const int colV  = h_i * DKH + sl * DVS;

  const v8f vz = {0.f,0.f,0.f,0.f,0.f,0.f,0.f,0.f};

  for (int i = tid; i < DKH * DVS; i += 256) sS[(i / DVS) * LDT + (i % DVS)] = 0.f;
  __syncthreads();

  for (int c = 0; c < L / CH; ++c) {
    const int t0 = c * CH;
    // load q,k chunk (vectorized b128) + beta
    for (int i = tid; i < CH * DKH / 4; i += 256) {
      int row = i >> 6, col = (i & 63) << 2;
      size_t g = (rowBase + t0 + row) * D + colQK + col;
      *(v4f*)&sQ[row * LDQ + col] = *(const v4f*)&qg[g];
      *(v4f*)&sK[row * LDQ + col] = *(const v4f*)&kg[g];
    }
    if (tid < CH) sBt[tid] = betag[(rowBase + t0 + tid) * NHEADS + h_i];
    __syncthreads();
    // row l2 norms
    if (tid < CH) {
      float aq = 0.f, ak = 0.f;
      for (int j = 0; j < DKH; ++j) {
        float x = sQ[tid * LDQ + j]; aq += x * x;
        float y = sK[tid * LDQ + j]; ak += y * y;
      }
      sNq[tid] = rsqrtf(aq + 1e-6f);
      sNk[tid] = rsqrtf(ak + 1e-6f);
    }
    __syncthreads();
    for (int i = tid; i < CH * DKH; i += 256) {
      int row = i >> 8, col = i & 255;
      float qn = sQ[row * LDQ + col] * sNq[row];
      float kn = sK[row * LDQ + col] * sNk[row];
      sQ [row * LDQ + col] = qn;
      sK [row * LDQ + col] = kn;
      sKB[row * LDQ + col] = kn * sBt[row];
    }
    for (int i = tid; i < CH * DVS / 4; i += 256) {
      int row = i >> 3, col = (i & 7) << 2;
      v4f vv = *(const v4f*)&vg[(rowBase + t0 + row) * D + colV + col];
      vv *= sBt[row];
      *(v4f*)&sV[row * LDT + col] = vv;
    }
    __syncthreads();
    // attn = strict_lower(-(KB @ K^T))  -> sT
    if (wave < 4) {
      const int m0 = ((wave >> 1) & 1) * 16, n0 = (wave & 1) * 16;
      v8f acc = vz;
      for (int k0 = 0; k0 < DKH; k0 += 4) {
        v2f a = { sKB[(m0 + r) * LDQ + k0 + half * 2], sKB[(m0 + r) * LDQ + k0 + half * 2 + 1] };
        v2f b = { sK [(n0 + r) * LDQ + k0 + half * 2], sK [(n0 + r) * LDQ + k0 + half * 2 + 1] };
        acc = wmma_f32x4(a, b, acc);
      }
      for (int j = 0; j < 8; ++j) {
        int row = m0 + half * 8 + j, col = n0 + r;
        sT[row * LDT + col] = (col < row) ? -acc[j] : 0.f;
      }
    }
    __syncthreads();
    // forward substitution (reference's sequential triangular recurrence)
    for (int i = 1; i < CH; ++i) {
      float nv = 0.f;
      if (tid < i) {
        nv = sT[i * LDT + tid];
        for (int t = tid + 1; t < i; ++t) nv += sT[i * LDT + t] * sT[t * LDT + tid];
      }
      __syncthreads();
      if (tid < i) sT[i * LDT + tid] = nv;
      __syncthreads();
    }
    if (tid < CH) sT[tid * LDT + tid] = 1.f;
    __syncthreads();
    // W = T @ KB (32 tiles / 8 waves) ; U = T @ V (waves 0-3)
    for (int tt = wave; tt < 32; tt += 8) {
      const int m0 = (tt & 1) * 16, n0 = (tt >> 1) * 16;
      v8f acc = vz;
      for (int k0 = 0; k0 < CH; k0 += 4) {
        v2f a = { sT [(m0 + r) * LDT + k0 + half * 2], sT [(m0 + r) * LDT + k0 + half * 2 + 1] };
        v2f b = { sKB[(k0 + half * 2) * LDQ + n0 + r], sKB[(k0 + half * 2 + 1) * LDQ + n0 + r] };
        acc = wmma_f32x4(a, b, acc);
      }
      for (int j = 0; j < 8; ++j) sW[(m0 + half * 8 + j) * LDQ + n0 + r] = acc[j];
    }
    if (wave < 4) {
      const int m0 = ((wave >> 1) & 1) * 16, n0 = (wave & 1) * 16;
      v8f acc = vz;
      for (int k0 = 0; k0 < CH; k0 += 4) {
        v2f a = { sT[(m0 + r) * LDT + k0 + half * 2], sT[(m0 + r) * LDT + k0 + half * 2 + 1] };
        v2f b = { sV[(k0 + half * 2) * LDT + n0 + r], sV[(k0 + half * 2 + 1) * LDT + n0 + r] };
        acc = wmma_f32x4(a, b, acc);
      }
      for (int j = 0; j < 8; ++j) sU[(m0 + half * 8 + j) * LDT + n0 + r] = acc[j];
    }
    __syncthreads();
    // u_adj = U - W@S (waves 0-3) || Ai = tril(Q@K^T) (waves 4-7)
    if (wave < 4) {
      const int m0 = ((wave >> 1) & 1) * 16, n0 = (wave & 1) * 16;
      v8f acc = vz;
      for (int k0 = 0; k0 < DKH; k0 += 4) {
        v2f a = { sW[(m0 + r) * LDQ + k0 + half * 2], sW[(m0 + r) * LDQ + k0 + half * 2 + 1] };
        v2f b = { sS[(k0 + half * 2) * LDT + n0 + r], sS[(k0 + half * 2 + 1) * LDT + n0 + r] };
        acc = wmma_f32x4(a, b, acc);
      }
      for (int j = 0; j < 8; ++j) sU[(m0 + half * 8 + j) * LDT + n0 + r] -= acc[j];
    } else {
      const int w2 = wave - 4;
      const int m0 = ((w2 >> 1) & 1) * 16, n0 = (w2 & 1) * 16;
      v8f acc = vz;
      for (int k0 = 0; k0 < DKH; k0 += 4) {
        v2f a = { sQ[(m0 + r) * LDQ + k0 + half * 2], sQ[(m0 + r) * LDQ + k0 + half * 2 + 1] };
        v2f b = { sK[(n0 + r) * LDQ + k0 + half * 2], sK[(n0 + r) * LDQ + k0 + half * 2 + 1] };
        acc = wmma_f32x4(a, b, acc);
      }
      for (int j = 0; j < 8; ++j) {
        int row = m0 + half * 8 + j, col = n0 + r;
        sAi[row * LDT + col] = (col <= row) ? acc[j] : 0.f;
      }
    }
    __syncthreads();
    // P = Ai @ u_adj (waves 4-7, into sT which is free now)
    if (wave >= 4) {
      const int w2 = wave - 4;
      const int m0 = ((w2 >> 1) & 1) * 16, n0 = (w2 & 1) * 16;
      v8f acc = vz;
      for (int k0 = 0; k0 < CH; k0 += 4) {
        v2f a = { sAi[(m0 + r) * LDT + k0 + half * 2], sAi[(m0 + r) * LDT + k0 + half * 2 + 1] };
        v2f b = { sU [(k0 + half * 2) * LDT + n0 + r], sU [(k0 + half * 2 + 1) * LDT + n0 + r] };
        acc = wmma_f32x4(a, b, acc);
      }
      for (int j = 0; j < 8; ++j) sT[(m0 + half * 8 + j) * LDT + n0 + r] = acc[j];
    }
    __syncthreads();
    // o = Q@S + P -> global (waves 0-3)
    if (wave < 4) {
      const int m0 = ((wave >> 1) & 1) * 16, n0 = (wave & 1) * 16;
      v8f acc = vz;
      for (int k0 = 0; k0 < DKH; k0 += 4) {
        v2f a = { sQ[(m0 + r) * LDQ + k0 + half * 2], sQ[(m0 + r) * LDQ + k0 + half * 2 + 1] };
        v2f b = { sS[(k0 + half * 2) * LDT + n0 + r], sS[(k0 + half * 2 + 1) * LDT + n0 + r] };
        acc = wmma_f32x4(a, b, acc);
      }
      for (int j = 0; j < 8; ++j) {
        int row = m0 + half * 8 + j, col = n0 + r;
        og[(rowBase + t0 + row) * D + colV + col] = acc[j] + sT[row * LDT + col];
      }
    }
    __syncthreads();
    // S += K^T @ u_adj (32 tiles / 8 waves, accumulate C directly from LDS)
    for (int tt = wave; tt < 32; tt += 8) {
      const int m0 = (tt >> 1) * 16, n0 = (tt & 1) * 16;
      v8f acc;
      for (int j = 0; j < 8; ++j) acc[j] = sS[(m0 + half * 8 + j) * LDT + n0 + r];
      for (int k0 = 0; k0 < CH; k0 += 4) {
        v2f a = { sK[(k0 + half * 2) * LDQ + m0 + r], sK[(k0 + half * 2 + 1) * LDQ + m0 + r] };
        v2f b = { sU[(k0 + half * 2) * LDT + n0 + r], sU[(k0 + half * 2 + 1) * LDT + n0 + r] };
        acc = wmma_f32x4(a, b, acc);
      }
      for (int j = 0; j < 8; ++j) sS[(m0 + half * 8 + j) * LDT + n0 + r] = acc[j];
    }
    __syncthreads();
  }
}

// ---------------- EMA scan over t per (b,h,j) ----------------
__global__ void ema_kernel(const float* __restrict__ v, const float* __restrict__ gamma,
                           float* __restrict__ out, int B, int L, int H, int dv) {
  int idx = blockIdx.x * blockDim.x + threadIdx.x;
  if (idx >= B * H * dv) return;
  int j = idx % dv, h = (idx / dv) % H, b = idx / (dv * H);
  float s = 0.f;
  for (int t = 0; t < L; ++t) {
    size_t rr = (size_t)b * L + t;
    float g  = gamma[rr * H + h];
    float vv = v[rr * (H * dv) + (size_t)h * dv + j];
    s = g * s + (1.f - g) * vv;
    out[rr * (H * dv) + (size_t)h * dv + j] = s;
  }
}

// ---------------- mix gate + per-head RMS norm -> bf16 ----------------
__global__ __launch_bounds__(256)
void mix_rms(const float* __restrict__ od, const float* __restrict__ ema,
             const float* __restrict__ mixinp, const float* __restrict__ mix_bias,
             const float* __restrict__ rmsw, __bf16* __restrict__ out, int H, int dv) {
  __shared__ float red[NHEADS * 256];
  int row = blockIdx.x, t = threadIdx.x;
  float vals[NHEADS];
  for (int h = 0; h < H; ++h) {
    float m = sigmoidf_(mixinp[row * H + h] + mix_bias[h]);
    size_t ix = (size_t)row * (H * dv) + (size_t)h * dv + t;
    float v = (1.f - m) * od[ix] + m * ema[ix];
    vals[h] = v;
    red[h * 256 + t] = v * v;
  }
  __syncthreads();
  for (int s = 128; s > 0; s >>= 1) {
    if (t < s)
      for (int h = 0; h < H; ++h) red[h * 256 + t] += red[h * 256 + t + s];
    __syncthreads();
  }
  for (int h = 0; h < H; ++h) {
    float scale = rsqrtf(red[h * 256] / (float)dv + 1e-5f);
    out[(size_t)row * (H * dv) + (size_t)h * dv + t] = (__bf16)(vals[h] * scale * rmsw[t]);
  }
}

// ---------------- host launch ----------------
extern "C" void kernel_launch(void* const* d_in, const int* in_sizes, int n_in,
                              void* d_out, int out_size, void* d_ws, size_t ws_size,
                              hipStream_t stream) {
  (void)in_sizes; (void)n_in; (void)out_size; (void)ws_size;
  const float* hidden   = (const float*)d_in[0];
  const float* Wq       = (const float*)d_in[1];
  const float* Wk       = (const float*)d_in[2];
  const float* Wv       = (const float*)d_in[3];
  const float* conv_q   = (const float*)d_in[4];
  const float* conv_k   = (const float*)d_in[5];
  const float* conv_v   = (const float*)d_in[6];
  const float* Wb       = (const float*)d_in[7];
  const float* Wdec     = (const float*)d_in[8];
  const float* Wmix     = (const float*)d_in[9];
  const float* conv_mix = (const float*)d_in[10];
  const float* mix_bias = (const float*)d_in[11];
  const float* rms_w    = (const float*)d_in[12];
  const float* Wo       = (const float*)d_in[13];

  const int B = BB, L = LL, D = DD, H = NHEADS, DV = DKH;
  const int M = B * L;       // 8192
  const int N3 = 3 * D;      // fused q|k|v projection width

  char* ws = (char*)d_ws;
  size_t off = 0;
  auto alloc = [&](size_t bytes) -> void* {
    void* p = ws + off;
    off += (bytes + 255) & ~(size_t)255;
    return p;
  };
  __bf16* hbf  = (__bf16*)alloc((size_t)M * D * 2);    // bf16 activations; reused for o_final
  __bf16* wt3  = (__bf16*)alloc((size_t)N3 * D * 2);   // transposed bf16 weights [N,K]; reused for Wo
  float*  lin3 = (float*) alloc((size_t)M * N3 * 4);   // fused GEMM out; reused as o_d
  float*  qb   = (float*) alloc((size_t)M * D * 4);
  float*  kb   = (float*) alloc((size_t)M * D * 4);
  float*  vb   = (float*) alloc((size_t)M * D * 4);
  float*  emab = (float*) alloc((size_t)M * D * 4);
  float*  beta = (float*) alloc((size_t)M * H * 4);
  float*  gamm = (float*) alloc((size_t)M * H * 4);
  float*  mixl = (float*) alloc((size_t)M * H * 4);
  float*  mixi = (float*) alloc((size_t)M * H * 4);

  const int TPB = 256;

  // hidden -> bf16 (vectorized)
  cvt_bf16_v8<<<(M * D / 8 + TPB - 1) / TPB, TPB, 0, stream>>>(hidden, hbf, M * D);
  // beta / gamma / mix_lin (L2-resident repeated reads of hidden)
  proj_small<<<(M * H * 3 + TPB - 1) / TPB, TPB, 0, stream>>>(hidden, Wb, Wdec, Wmix,
                                                              beta, gamm, mixl, M, D, H);
  // mix_inp = silu(conv(mix_lin))
  conv_silu<<<(B * L * H + TPB - 1) / TPB, TPB, 0, stream>>>(mixl, conv_mix, mixi,
                                                             B, L, H, H, 0, 1);

  // pack Wq|Wk|Wv transposed -> wt3 [3072 x 1024] bf16
  const int wblk = (D * D + TPB - 1) / TPB;
  cvt_transpose_bf16<<<wblk, TPB, 0, stream>>>(Wq, wt3 + (size_t)0 * D * D, D, D);
  cvt_transpose_bf16<<<wblk, TPB, 0, stream>>>(Wk, wt3 + (size_t)1 * D * D, D, D);
  cvt_transpose_bf16<<<wblk, TPB, 0, stream>>>(Wv, wt3 + (size_t)2 * D * D, D, D);

  // fused projection GEMM: [8192,1024] @ [1024,3072] -> lin3
  dim3 gg3(N3 / 256, M / 32);
  gemm_bf16_wmma<<<gg3, TPB, 0, stream>>>(hbf, wt3, lin3, M, N3, D);

  // conv + silu per projection slice
  conv_silu<<<(M * D + TPB - 1) / TPB, TPB, 0, stream>>>(lin3, conv_q, qb, B, L, D, N3, 0 * D, 2);
  conv_silu<<<(M * D + TPB - 1) / TPB, TPB, 0, stream>>>(lin3, conv_k, kb, B, L, D, N3, 1 * D, 2);
  conv_silu<<<(M * D + TPB - 1) / TPB, TPB, 0, stream>>>(lin3, conv_v, vb, B, L, D, N3, 2 * D, 1);

  // chunkwise delta rule -> o_d (reuse lin3 as [M x 1024])
  float* od = lin3;
  const int shbytes = (4 * CH * LDQ + DKH * LDT + 4 * CH * LDT + 3 * CH) * 4;  // ~184.4 KB
  (void)hipFuncSetAttribute((const void*)delta_kernel,
                            hipFuncAttributeMaxDynamicSharedMemorySize, shbytes);
  delta_kernel<<<B * H * (DV / DVS), 256, shbytes, stream>>>(qb, kb, vb, beta, od, B, L);

  // EMA scan
  ema_kernel<<<(B * H * DV + TPB - 1) / TPB, TPB, 0, stream>>>(vb, gamm, emab, B, L, H, DV);

  // mix + rms norm -> bf16 (reuse hbf)
  mix_rms<<<M, 256, 0, stream>>>(od, emab, mixi, mix_bias, rms_w, hbf, H, DV);

  // out = o_final @ Wo -> d_out (f32); reuse wt3 for transposed Wo
  cvt_transpose_bf16<<<wblk, TPB, 0, stream>>>(Wo, wt3, D, D);
  dim3 gg1(D / 256, M / 32);
  gemm_bf16_wmma<<<gg1, TPB, 0, stream>>>(hbf, wt3, (float*)d_out, M, D, D);
}
